// FraudDetectionHybrid_65481071409833
// MI455X (gfx1250) — matrix operand order repaired and checked by reading, not compiled, not run on hardware
//
#include <hip/hip_runtime.h>

typedef unsigned short u16;
typedef unsigned int   u32;

typedef __attribute__((ext_vector_type(16))) __bf16 v16bf;
typedef __attribute__((ext_vector_type(8)))  float  v8f;

#define B_TOTAL 65536
#define F_IN    256
#define D_IN    257      // 256 raw + 1 conv feature
#define AKPAD   288      // K of layer-1 padded to multiple of 32
#define ASTRIDE 296      // LDS row stride (halves) for A tile  (bank-conflict pad)
#define H1      1024
#define H1STR   1032     // LDS row stride (halves) for H1 tile
#define H2      512
#define H2STR   520      // LDS row stride (halves) for H2 tile
#define ROWS    64       // rows per workgroup (= 4 WMMA m-tiles)

#define LDS_BYTES (ROWS*H1STR*2 + ROWS*H2STR*2)   // 132096 + 66560 = 198656 B

__device__ __forceinline__ u16 f2bf(float f) {
    u32 u = __float_as_uint(f);
    u32 r = (u + 0x7FFFu + ((u >> 16) & 1u)) >> 16;   // round-to-nearest-even
    return (u16)r;
}
__device__ __forceinline__ float bf2f(u32 h) {
    return __uint_as_float(h << 16);
}

// ---------------------------------------------------------------------------
// Pack kernel: W1 -> bf16 [1024][288] (transposed, zero-padded K),
//              W2 -> bf16 [512][1024] (transposed),
//              BN folded into per-feature scale/shift: s=g*rsqrt(v+eps),
//              t = s*(b - m) + beta   (so layer = relu(s*(x@W) + t))
// ---------------------------------------------------------------------------
__global__ void __launch_bounds__(256)
pack_kernel(const float* __restrict__ W1, const float* __restrict__ b1,
            const float* __restrict__ g1, const float* __restrict__ be1,
            const float* __restrict__ m1, const float* __restrict__ v1,
            const float* __restrict__ W2, const float* __restrict__ b2,
            const float* __restrict__ g2, const float* __restrict__ be2,
            const float* __restrict__ m2, const float* __restrict__ v2,
            u16* __restrict__ W1p, u16* __restrict__ W2p,
            float* __restrict__ s1, float* __restrict__ t1,
            float* __restrict__ s2, float* __restrict__ t2) {
    const int N1 = H1 * AKPAD;      // 294912
    const int N2 = H2 * H1;         // 524288
    int idx = blockIdx.x * 256 + threadIdx.x;
    if (idx < N1) {
        int n = idx / AKPAD, k = idx - n * AKPAD;
        W1p[idx] = (k < D_IN) ? f2bf(W1[k * H1 + n]) : (u16)0;
    } else if (idx < N1 + N2) {
        int j = idx - N1;
        int n = j >> 10, k = j & 1023;
        W2p[j] = f2bf(W2[k * H2 + n]);
    } else if (idx < N1 + N2 + H1) {
        int i = idx - N1 - N2;
        float s = g1[i] * rsqrtf(v1[i] + 1e-5f);
        s1[i] = s;
        t1[i] = s * (b1[i] - m1[i]) + be1[i];
    } else if (idx < N1 + N2 + H1 + H2) {
        int i = idx - N1 - N2 - H1;
        float s = g2[i] * rsqrtf(v2[i] + 1e-5f);
        s2[i] = s;
        t2[i] = s * (b2[i] - m2[i]) + be2[i];
    }
}

// ---------------------------------------------------------------------------
// Fused MLP: per-WG 64-row tile, bf16 WMMA (f32 accumulate) through both GEMMs.
// Register tile per wave = 4 m-tiles x 2 n-tiles (C = 64 VGPRs):
//   each global B-fragment feeds 4 WMMAs, each LDS A-fragment feeds 2 WMMAs.
// ---------------------------------------------------------------------------
__global__ void __launch_bounds__(256)
fused_mlp_kernel(const float* __restrict__ x,
                 const u16* __restrict__ W1p, const u16* __restrict__ W2p,
                 const float* __restrict__ s1, const float* __restrict__ t1,
                 const float* __restrict__ s2, const float* __restrict__ t2,
                 const float* __restrict__ W3, const float* __restrict__ b3,
                 float* __restrict__ out) {
    extern __shared__ u16 smem[];
    u16* H1b = smem;                       // [ROWS][H1STR]  (129 KB)
    u16* Ab  = smem + ROWS * H1STR;        // [ROWS][ASTRIDE] overlaid with...
    u16* H2b = smem + ROWS * H1STR;        // [ROWS][H2STR]  (A is dead by then)

    const int t    = threadIdx.x;
    const int lane = t & 31;
    const int wave = t >> 5;               // 0..7
    const int g    = lane >> 4;            // half-wave id
    const int l16  = lane & 15;

    // ---------------- Stage 0: x tile -> mean/conv feature -> bf16 A in LDS
    {
        u32* Az = (u32*)Ab;
        const int AW = ROWS * ASTRIDE / 2;
        for (int i = t; i < AW; i += 256) Az[i] = 0u;   // zero incl. K padding
        __syncthreads();

        const int r = t >> 2, q = t & 3;                // 4 threads per row
        const long grow = (long)blockIdx.x * ROWS + r;
        const float* xp = x + grow * F_IN + q * 64;
        u16* arow = Ab + r * ASTRIDE + q * 64;
        float sum = 0.f;
#pragma unroll
        for (int i = 0; i < 64; i += 4) {
            float4 v = *(const float4*)(xp + i);
            sum += v.x + v.y + v.z + v.w;
            *(u32*)(arow + i)     = (u32)f2bf(v.x) | ((u32)f2bf(v.y) << 16);
            *(u32*)(arow + i + 2) = (u32)f2bf(v.z) | ((u32)f2bf(v.w) << 16);
        }
        sum += __shfl_xor(sum, 1, 32);
        sum += __shfl_xor(sum, 2, 32);
        if (q == 0) {
            float mean = sum * (1.0f / 256.0f);
            Ab[r * ASTRIDE + 256] = f2bf(mean > 0.0f ? 1.0f : 0.0f);
        }
    }
    __syncthreads();

    // ---------------- Stage 1: H1 = relu(s1*(A @ W1) + t1), bf16 into LDS
    // 64 n-tiles: each wave takes 4 pairs of adjacent n-tiles, all 4 m-tiles.
#pragma unroll 1
    for (int ip = 0; ip < 4; ++ip) {
        const int nt0 = ip * 16 + wave * 2;            // n-tiles nt0, nt0+1
        const u16* brow0 = W1p + (nt0 * 16 + l16) * AKPAD;
        const u16* brow1 = brow0 + 16 * AKPAD;
        const u16* abase = Ab + l16 * ASTRIDE;
        v8f c0[4] = {}, c1[4] = {};
#pragma unroll
        for (int k = 0; k < AKPAD; k += 32) {
            v16bf b0, b1;
            uint4* bp0 = (uint4*)&b0;
            bp0[0] = *(const uint4*)(brow0 + k + 16 * g);    // B: 16 contiguous K @ 16g
            bp0[1] = *(const uint4*)(brow0 + k + 16 * g + 8);
            uint4* bp1 = (uint4*)&b1;
            bp1[0] = *(const uint4*)(brow1 + k + 16 * g);
            bp1[1] = *(const uint4*)(brow1 + k + 16 * g + 8);
#pragma unroll
            for (int m = 0; m < 4; ++m) {
                v16bf a;
                uint4* ap = (uint4*)&a;
                const u16* ar = abase + m * 16 * ASTRIDE + k + 16 * g;
                ap[0] = *(const uint4*)(ar);                 // VGPR0..3: K = 16g..16g+7
                ap[1] = *(const uint4*)(ar + 8);             // VGPR4..7: K = 16g+8..
                c0[m] = __builtin_amdgcn_wmma_f32_16x16x32_bf16(
                            false, a, false, b0, (short)0, c0[m], false, false);
                c1[m] = __builtin_amdgcn_wmma_f32_16x16x32_bf16(
                            false, a, false, b1, (short)0, c1[m], false, false);
            }
        }
        const int col0 = nt0 * 16 + l16, col1 = col0 + 16;
        const float sa = s1[col0], ta = t1[col0];
        const float sb = s1[col1], tb = t1[col1];
#pragma unroll
        for (int m = 0; m < 4; ++m) {
            u16* hb = H1b + (m * 16 + 8 * g) * H1STR;        // C: row = 8g+j
#pragma unroll
            for (int j = 0; j < 8; ++j) {
                hb[j * H1STR + col0] = f2bf(fmaxf(sa * c0[m][j] + ta, 0.0f));
                hb[j * H1STR + col1] = f2bf(fmaxf(sb * c1[m][j] + tb, 0.0f));
            }
        }
    }
    __syncthreads();

    // ---------------- Stage 2: H2 = relu(s2*(H1 @ W2) + t2), bf16 into LDS
    // 32 n-tiles: each wave takes 2 pairs of adjacent n-tiles, all 4 m-tiles.
#pragma unroll 1
    for (int ip = 0; ip < 2; ++ip) {
        const int nt0 = ip * 16 + wave * 2;
        const u16* brow0 = W2p + (nt0 * 16 + l16) * H1;
        const u16* brow1 = brow0 + 16 * H1;
        const u16* abase = H1b + l16 * H1STR;
        v8f c0[4] = {}, c1[4] = {};
#pragma unroll 2
        for (int k = 0; k < H1; k += 32) {
            v16bf b0, b1;
            uint4* bp0 = (uint4*)&b0;
            bp0[0] = *(const uint4*)(brow0 + k + 16 * g);
            bp0[1] = *(const uint4*)(brow0 + k + 16 * g + 8);
            uint4* bp1 = (uint4*)&b1;
            bp1[0] = *(const uint4*)(brow1 + k + 16 * g);
            bp1[1] = *(const uint4*)(brow1 + k + 16 * g + 8);
#pragma unroll
            for (int m = 0; m < 4; ++m) {
                v16bf a;
                uint4* ap = (uint4*)&a;
                const u16* ar = abase + m * 16 * H1STR + k + 16 * g;
                ap[0] = *(const uint4*)(ar);
                ap[1] = *(const uint4*)(ar + 8);
                c0[m] = __builtin_amdgcn_wmma_f32_16x16x32_bf16(
                            false, a, false, b0, (short)0, c0[m], false, false);
                c1[m] = __builtin_amdgcn_wmma_f32_16x16x32_bf16(
                            false, a, false, b1, (short)0, c1[m], false, false);
            }
        }
        const int col0 = nt0 * 16 + l16, col1 = col0 + 16;
        const float sa = s2[col0], ta = t2[col0];
        const float sb = s2[col1], tb = t2[col1];
#pragma unroll
        for (int m = 0; m < 4; ++m) {
            u16* hb = H2b + (m * 16 + 8 * g) * H2STR;
#pragma unroll
            for (int j = 0; j < 8; ++j) {
                hb[j * H2STR + col0] = f2bf(fmaxf(sa * c0[m][j] + ta, 0.0f));
                hb[j * H2STR + col1] = f2bf(fmaxf(sb * c1[m][j] + tb, 0.0f));
            }
        }
    }
    __syncthreads();

    // ---------------- Stage 3: out = sigmoid(H2 . W3 + b3), 4 lanes per row
    {
        const int r = t >> 2, q = t & 3;
        const long grow = (long)blockIdx.x * ROWS + r;
        const u16* hr = H2b + r * H2STR + q * 128;
        const float* w3 = W3 + q * 128;
        float sum = 0.f;
#pragma unroll 4
        for (int i = 0; i < 128; i += 4) {
            u32 h01 = *(const u32*)(hr + i);
            u32 h23 = *(const u32*)(hr + i + 2);
            float4 w = *(const float4*)(w3 + i);
            sum += bf2f(h01 & 0xFFFFu) * w.x + bf2f(h01 >> 16) * w.y
                 + bf2f(h23 & 0xFFFFu) * w.z + bf2f(h23 >> 16) * w.w;
        }
        sum += __shfl_xor(sum, 1, 32);
        sum += __shfl_xor(sum, 2, 32);
        if (q == 0) {
            float z = sum + b3[0];
            out[grow] = 1.0f / (1.0f + expf(-z));
        }
    }
}

// ---------------------------------------------------------------------------
extern "C" void kernel_launch(void* const* d_in, const int* in_sizes, int n_in,
                              void* d_out, int out_size, void* d_ws, size_t ws_size,
                              hipStream_t stream) {
    const float* x   = (const float*)d_in[0];
    const float* W1  = (const float*)d_in[1];
    const float* b1  = (const float*)d_in[2];
    const float* g1  = (const float*)d_in[3];
    const float* be1 = (const float*)d_in[4];
    const float* m1  = (const float*)d_in[5];
    const float* v1  = (const float*)d_in[6];
    const float* W2  = (const float*)d_in[7];
    const float* b2  = (const float*)d_in[8];
    const float* g2  = (const float*)d_in[9];
    const float* be2 = (const float*)d_in[10];
    const float* m2  = (const float*)d_in[11];
    const float* v2  = (const float*)d_in[12];
    const float* W3  = (const float*)d_in[13];
    const float* b3  = (const float*)d_in[14];

    char* ws = (char*)d_ws;
    u16*   W1p = (u16*)ws;                                   // 1024*288*2 = 589824 B
    u16*   W2p = (u16*)(ws + 589824);                        // 512*1024*2 = 1048576 B
    float* s1  = (float*)(ws + 589824 + 1048576);
    float* t1  = s1 + H1;
    float* s2  = t1 + H1;
    float* t2  = s2 + H2;

    const int packN = H1 * AKPAD + H2 * H1 + H1 + H2;        // 820736
    pack_kernel<<<(packN + 255) / 256, 256, 0, stream>>>(
        W1, b1, g1, be1, m1, v1, W2, b2, g2, be2, m2, v2,
        W1p, W2p, s1, t1, s2, t2);

    fused_mlp_kernel<<<B_TOTAL / ROWS, 256, LDS_BYTES, stream>>>(
        x, W1p, W2p, s1, t1, s2, t2, W3, b3, (float*)d_out);
}
